// CrossModalAttention_79809082294793
// MI455X (gfx1250) — matrix operand order repaired and checked by reading, not compiled, not run on hardware
//
#include <hip/hip_runtime.h>
#include <hip/hip_bf16.h>

typedef unsigned short u16;
typedef __bf16 v16bf __attribute__((ext_vector_type(16)));
typedef float  v8f  __attribute__((ext_vector_type(8)));

#define NPOS 4096   // H*W
#define EMB  128
#define CIN  256
#define BATCH 8

union BFPack { uint4 u[2]; v16bf v; };

// A-operand: per-lane row = lane&15; elements 0..15 = K[k0..k0+15]
// caller passes k0 = chunk_base + h*16   (h = lane>>4)
__device__ __forceinline__ v16bf load_a(const u16* base, int k0) {
  BFPack p;
  p.u[0] = *(const uint4*)(base + k0);
  p.u[1] = *(const uint4*)(base + k0 + 8);
  return p.v;
}
// B-operand: per-lane col = lane&15; elements = 16 contiguous K starting at chunk_base + h*16
__device__ __forceinline__ v16bf load_b(const u16* base) {
  BFPack p;
  p.u[0] = *(const uint4*)(base);
  p.u[1] = *(const uint4*)(base + 8);
  return p.v;
}

__device__ __forceinline__ v8f wmma_bf16(v16bf a, v16bf b, v8f c) {
  return __builtin_amdgcn_wmma_f32_16x16x32_bf16(false, a, false, b, (short)0, c, false, false);
}

__device__ __forceinline__ u16 f2bf(float x) {
  unsigned u = __float_as_uint(x);
  u += 0x7FFFu + ((u >> 16) & 1u);          // round-to-nearest-even
  return (u16)(u >> 16);
}

__device__ __forceinline__ float rowmax16(float x) {
#pragma unroll
  for (int off = 1; off < 16; off <<= 1) x = fmaxf(x, __shfl_xor(x, off, 32));
  return x;
}
__device__ __forceinline__ float rowsum16(float x) {
#pragma unroll
  for (int off = 1; off < 16; off <<= 1) x += __shfl_xor(x, off, 32);
  return x;
}

// ---------------- prep: weights -> bf16, BN fold ----------------
__global__ void prep_kernel(const float* w0, const float* w1, const float* w2,
                            const float* w3, const float* w4, const float* w5,
                            const float* wp, const float* g, const float* be,
                            const float* mu, const float* va,
                            u16* wbf, u16* wpbf, float* bninv, float* bnsh) {
  int tid = blockIdx.x * 256 + threadIdx.x;
  int stride = gridDim.x * 256;
  const float* ws[6] = {w0, w1, w2, w3, w4, w5};
  for (int i = tid; i < 6 * EMB * CIN; i += stride)
    wbf[i] = f2bf(ws[i >> 15][i & 32767]);
  for (int i = tid; i < 256 * 256; i += stride)
    wpbf[i] = f2bf(wp[i]);
  if (tid < 256) {
    float iv = g[tid] * rsqrtf(va[tid] + 1e-5f);
    bninv[tid] = iv;
    bnsh[tid]  = be[tid] - mu[tid] * iv;
  }
}

// ---------------- transpose: (B,C,N) f32 -> (B,N,C) bf16 ----------------
__global__ void transpose_kernel(const float* __restrict__ in, u16* __restrict__ out) {
  size_t idx = (size_t)blockIdx.x * 256 + threadIdx.x;   // B*N*C threads
  int c = (int)(idx & (CIN - 1));
  size_t bn = idx >> 8;
  int n = (int)(bn & (NPOS - 1));
  int b = (int)(bn >> 12);
  out[idx] = f2bf(in[((size_t)b * CIN + c) * NPOS + n]);
}

// ---------------- projection: (N,C) x W^T + bias -> (N,E) bf16 ----------------
__launch_bounds__(256)
__global__ void proj_kernel(const u16* __restrict__ fnc, const u16* __restrict__ wbf,
                            const float* __restrict__ bias, u16* __restrict__ out) {
  int tid = threadIdx.x, wave = tid >> 5, lane = tid & 31;
  int col = lane & 15, h = lane >> 4;
  int b = blockIdx.y;
  int n0 = (blockIdx.x * 8 + wave) * 16;

  const u16* frow = fnc + ((size_t)b * NPOS + n0 + col) * CIN;
  v16bf a[8];
#pragma unroll
  for (int cc = 0; cc < 8; ++cc) a[cc] = load_a(frow, cc * 32 + h * 16);

#pragma unroll
  for (int et = 0; et < 8; ++et) {
    v8f acc = {0.f, 0.f, 0.f, 0.f, 0.f, 0.f, 0.f, 0.f};
    const u16* wrow = wbf + (size_t)(et * 16 + col) * CIN;
#pragma unroll
    for (int cc = 0; cc < 8; ++cc)
      acc = wmma_bf16(a[cc], load_b(wrow + cc * 32 + h * 16), acc);
    float bv = bias[et * 16 + col];
#pragma unroll
    for (int j = 0; j < 8; ++j) {
      int n = n0 + j + 8 * h;
      out[((size_t)b * NPOS + n) * EMB + et * 16 + col] = f2bf(acc[j] + bv);
    }
  }
}

// ---------------- flash attention: softmax(QK^T*s)V -> comb (B,N,256) bf16 ------
__launch_bounds__(256)
__global__ void attn_kernel(const u16* __restrict__ Q, const u16* __restrict__ K,
                            const u16* __restrict__ V, u16* __restrict__ comb,
                            int dirOff) {
  const int tid = threadIdx.x, wave = tid >> 5, lane = tid & 31;
  const int col = lane & 15, h = lane >> 4;
  const int b = blockIdx.y;
  const int n0 = (blockIdx.x * 8 + wave) * 16;

  const u16* Qb = Q + (size_t)b * NPOS * EMB;
  const u16* Kb = K + (size_t)b * NPOS * EMB;
  const u16* Vb = V + (size_t)b * NPOS * EMB;

  __shared__ __align__(16) u16 Klds[32 * 136];        // (m,e) padded
  __shared__ __align__(16) u16 Vtlds[128 * 40];       // (e,m) padded (transposed V)
  __shared__ __align__(16) u16 Plds[8 * 16 * 40];     // per-wave P tile (n,k)

  // preload Q rows (16 x 128) as A operands
  v16bf qa[4];
  {
    const u16* qrow = Qb + (size_t)(n0 + col) * EMB;
#pragma unroll
    for (int ec = 0; ec < 4; ++ec) qa[ec] = load_a(qrow, ec * 32 + h * 16);
  }

  v8f o[8];
#pragma unroll
  for (int t = 0; t < 8; ++t)
#pragma unroll
    for (int j = 0; j < 8; ++j) o[t][j] = 0.f;
  float m[8], l[8];
#pragma unroll
  for (int j = 0; j < 8; ++j) { m[j] = -1e30f; l[j] = 0.f; }

  const float scale = 0.08838834764831845f;   // 128^-0.5
  const int r = tid >> 3;                     // staging row 0..31
  const int e0 = (tid & 7) * 16;              // staging e offset

  for (int c0 = 0; c0 < NPOS; c0 += 32) {
    __syncthreads();   // previous iteration's LDS reads done
    // ---- stage K tile (32 x 128) and transposed V tile into LDS ----
    {
      const uint4* ksrc = (const uint4*)(Kb + (size_t)(c0 + r) * EMB + e0);
      uint4* kdst = (uint4*)(Klds + r * 136 + e0);
      kdst[0] = ksrc[0];
      kdst[1] = ksrc[1];
      const u16* vsrc = Vb + (size_t)(c0 + r) * EMB + e0;
#pragma unroll
      for (int i = 0; i < 16; ++i) Vtlds[(e0 + i) * 40 + r] = vsrc[i];
      if (c0 + 32 < NPOS) {   // prefetch next tiles
        __builtin_prefetch(Kb + (size_t)(c0 + 32 + r) * EMB + e0, 0, 1);
        __builtin_prefetch(Vb + (size_t)(c0 + 32 + r) * EMB + e0, 0, 1);
      }
    }
    __syncthreads();

    // ---- S = Q K^T for 2 column sub-tiles ----
    v8f s0 = {0.f,0.f,0.f,0.f,0.f,0.f,0.f,0.f};
    v8f s1 = {0.f,0.f,0.f,0.f,0.f,0.f,0.f,0.f};
#pragma unroll
    for (int ec = 0; ec < 4; ++ec) {
      s0 = wmma_bf16(qa[ec], load_b(Klds + col * 136 + ec * 32 + h * 16), s0);
      s1 = wmma_bf16(qa[ec], load_b(Klds + (16 + col) * 136 + ec * 32 + h * 16), s1);
    }

    // ---- online softmax update; write P to LDS in A layout ----
    u16* Pw = Plds + wave * 16 * 40;
#pragma unroll
    for (int j = 0; j < 8; ++j) {
      float v0 = s0[j] * scale, v1 = s1[j] * scale;
      float tm = rowmax16(fmaxf(v0, v1));
      float mn = fmaxf(m[j], tm);
      float al = __expf(m[j] - mn);
      float p0 = __expf(v0 - mn);
      float p1 = __expf(v1 - mn);
      float rs = rowsum16(p0 + p1);
      l[j] = l[j] * al + rs;
      m[j] = mn;
#pragma unroll
      for (int t = 0; t < 8; ++t) o[t][j] *= al;
      int row = j + 8 * h;
      Pw[row * 40 + col]      = f2bf(p0);
      Pw[row * 40 + 16 + col] = f2bf(p1);
    }
    __syncthreads();   // P visible (cross-lane within wave via LDS)

    // ---- O += P V ----
    v16bf pa = load_a(Pw + col * 40, h * 16);
#pragma unroll
    for (int t = 0; t < 8; ++t)
      o[t] = wmma_bf16(pa, load_b(Vtlds + (t * 16 + col) * 40 + h * 16), o[t]);
  }

  // ---- epilogue: divide by l, store to comb (B,N,256) ----
#pragma unroll
  for (int t = 0; t < 8; ++t) {
#pragma unroll
    for (int j = 0; j < 8; ++j) {
      int n = n0 + j + 8 * h;
      float val = o[t][j] / l[j];
      comb[((size_t)b * NPOS + n) * 256 + dirOff + t * 16 + col] = f2bf(val);
    }
  }
}

// ---------------- final: comb x w_proj^T, BN + ReLU -> (B,256,N) f32 -----------
__launch_bounds__(256)
__global__ void final_kernel(const u16* __restrict__ comb, const u16* __restrict__ wp,
                             const float* __restrict__ bninv, const float* __restrict__ bnsh,
                             float* __restrict__ out) {
  int tid = threadIdx.x, wave = tid >> 5, lane = tid & 31;
  int col = lane & 15, h = lane >> 4;
  int b = blockIdx.y;
  int n0 = blockIdx.x * 16;

  const u16* crow = comb + ((size_t)b * NPOS + n0 + col) * 256;
  v16bf a[8];
#pragma unroll
  for (int cc = 0; cc < 8; ++cc) a[cc] = load_a(crow, cc * 32 + h * 16);

#pragma unroll
  for (int sub = 0; sub < 2; ++sub) {
    int ot = wave * 2 + sub;
    v8f acc = {0.f, 0.f, 0.f, 0.f, 0.f, 0.f, 0.f, 0.f};
    const u16* wrow = wp + (size_t)(ot * 16 + col) * 256;
#pragma unroll
    for (int cc = 0; cc < 8; ++cc)
      acc = wmma_bf16(a[cc], load_b(wrow + cc * 32 + h * 16), acc);
    int oc = ot * 16 + col;
    float iv = bninv[oc], sh = bnsh[oc];
#pragma unroll
    for (int j = 0; j < 8; ++j) {
      int n = n0 + j + 8 * h;
      out[((size_t)b * 256 + oc) * NPOS + n] = fmaxf(acc[j] * iv + sh, 0.f);
    }
  }
}

extern "C" void kernel_launch(void* const* d_in, const int* in_sizes, int n_in,
                              void* d_out, int out_size, void* d_ws, size_t ws_size,
                              hipStream_t stream) {
  (void)in_sizes; (void)n_in; (void)out_size; (void)ws_size;
  char* ws = (char*)d_ws;
  size_t off = 0;
  auto alloc = [&](size_t bytes) { size_t o = off; off = (off + bytes + 255) & ~(size_t)255; return o; };

  u16*   wbf   = (u16*)(ws + alloc((size_t)6 * EMB * CIN * 2));
  u16*   wpbf  = (u16*)(ws + alloc((size_t)256 * 256 * 2));
  float* bninv = (float*)(ws + alloc(256 * 4));
  float* bnsh  = (float*)(ws + alloc(256 * 4));
  u16*   frgb  = (u16*)(ws + alloc((size_t)BATCH * NPOS * CIN * 2));
  u16*   fpl   = (u16*)(ws + alloc((size_t)BATCH * NPOS * CIN * 2));
  size_t qkvB  = (size_t)BATCH * NPOS * EMB * 2;
  u16* qrgb = (u16*)(ws + alloc(qkvB));
  u16* kpl  = (u16*)(ws + alloc(qkvB));
  u16* vpl  = (u16*)(ws + alloc(qkvB));
  u16* qpl  = (u16*)(ws + alloc(qkvB));
  u16* krgb = (u16*)(ws + alloc(qkvB));
  u16* vrgb = (u16*)(ws + alloc(qkvB));
  u16* comb = (u16*)(ws + alloc((size_t)BATCH * NPOS * 256 * 2));

  const float* F_RGB = (const float*)d_in[0];
  const float* F_PL  = (const float*)d_in[1];

  prep_kernel<<<64, 256, 0, stream>>>(
      (const float*)d_in[2], (const float*)d_in[4], (const float*)d_in[6],
      (const float*)d_in[8], (const float*)d_in[10], (const float*)d_in[12],
      (const float*)d_in[14], (const float*)d_in[15], (const float*)d_in[16],
      (const float*)d_in[17], (const float*)d_in[18],
      wbf, wpbf, bninv, bnsh);

  transpose_kernel<<<(BATCH * NPOS * CIN) / 256, 256, 0, stream>>>(F_RGB, frgb);
  transpose_kernel<<<(BATCH * NPOS * CIN) / 256, 256, 0, stream>>>(F_PL, fpl);

  dim3 pgrid(32, BATCH);
  proj_kernel<<<pgrid, 256, 0, stream>>>(frgb, wbf + 0 * EMB * CIN, (const float*)d_in[3],  qrgb);
  proj_kernel<<<pgrid, 256, 0, stream>>>(fpl,  wbf + 1 * EMB * CIN, (const float*)d_in[5],  kpl);
  proj_kernel<<<pgrid, 256, 0, stream>>>(fpl,  wbf + 2 * EMB * CIN, (const float*)d_in[7],  vpl);
  proj_kernel<<<pgrid, 256, 0, stream>>>(fpl,  wbf + 3 * EMB * CIN, (const float*)d_in[9],  qpl);
  proj_kernel<<<pgrid, 256, 0, stream>>>(frgb, wbf + 4 * EMB * CIN, (const float*)d_in[11], krgb);
  proj_kernel<<<pgrid, 256, 0, stream>>>(frgb, wbf + 5 * EMB * CIN, (const float*)d_in[13], vrgb);

  dim3 agrid(32, BATCH);
  attn_kernel<<<agrid, 256, 0, stream>>>(qrgb, kpl, vpl, comb, 0);
  attn_kernel<<<agrid, 256, 0, stream>>>(qpl, krgb, vrgb, comb, 128);

  dim3 fgrid(NPOS / 16, BATCH);
  final_kernel<<<fgrid, 256, 0, stream>>>(comb, wpbf, bninv, bnsh, (float*)d_out);
}
